// KANPyBlock_56083682951253
// MI455X (gfx1250) — matrix operand order repaired and checked by reading, not compiled, not run on hardware
//
#include <hip/hip_runtime.h>
#include <hip/hip_fp16.h>

typedef __attribute__((ext_vector_type(16))) _Float16 v16h;
typedef __attribute__((ext_vector_type(8)))  float    v8f;

#define IN_F   1024
#define OUT_F  1024
#define BATCH  4096
#define NB     8                      // NUM + K basis funcs per feature
#define GRIDP  12                     // NUM + 2K + 1 knots
#define KSPL   (IN_F * NB)            // 8192 spline K region
#define KDIM   (KSPL + IN_F)          // 9216 total GEMM K
#define LN_EPS 1e-5f

#define TILE_M 128
#define TILE_N 128
#define KSTEP  32
#define NSTEPS (KDIM / KSTEP)         // 288
#define APITCH 80                     // LDS bytes per 32-K row (64 data + 16 pad)
#define LDS_TILE (TILE_M * APITCH)    // 10240 B

// workspace layout (bytes)
#define WH_BYTES ((size_t)OUT_F * KDIM * 2)
#define WH_OFF   ((size_t)0)
#define WL_OFF   (WH_OFF + WH_BYTES)
#define Y_OFF    (WL_OFF + WH_BYTES)

// ---- CDNA5 async global->LDS path (guarded; falls back to reg pipeline) ----
#if __has_builtin(__builtin_amdgcn_global_load_async_to_lds_b128)
#define HAVE_ASYNC_LDS 1
typedef int v4i_vs __attribute__((vector_size(16)));
__device__ __forceinline__ void async_copy_b128(const void* g, void* l) {
  __builtin_amdgcn_global_load_async_to_lds_b128(
      (__attribute__((address_space(1))) v4i_vs*)g,
      (__attribute__((address_space(3))) v4i_vs*)l, 0, 0);
}
__device__ __forceinline__ void wait_async_keep4() {
#if __has_builtin(__builtin_amdgcn_s_wait_asynccnt)
  __builtin_amdgcn_s_wait_asynccnt(4);
#else
  asm volatile("s_wait_asynccnt 4" ::: "memory");
#endif
}
__device__ __forceinline__ void wait_async_all() {
#if __has_builtin(__builtin_amdgcn_s_wait_asynccnt)
  __builtin_amdgcn_s_wait_asynccnt(0);
#else
  asm volatile("s_wait_asynccnt 0" ::: "memory");
#endif
}
#else
#define HAVE_ASYNC_LDS 0
#endif

// ---------------------------------------------------------------------------
// Kernel 1: pack weights, transposed [o][K], split into fp16 hi + lo planes.
//   K layout: [i*8+k] = scale_sp[i,o]*coef[i,o,k]   (K < 8192)
//             [8192+i] = scale_base[i,o]
// ---------------------------------------------------------------------------
__global__ __launch_bounds__(256) void pack_w(
    const float* __restrict__ coef, const float* __restrict__ scale_base,
    const float* __restrict__ scale_sp,
    _Float16* __restrict__ wh, _Float16* __restrict__ wl) {
  int idx = blockIdx.x * blockDim.x + threadIdx.x;
  if (idx >= IN_F * OUT_F) return;
  int i = idx / OUT_F, o = idx % OUT_F;
  float ssp = scale_sp[(size_t)i * OUT_F + o];
  const float* c = coef + ((size_t)i * OUT_F + o) * NB;
  union { _Float16 e[8]; uint4 q; } ph, pl;
#pragma unroll
  for (int k = 0; k < NB; ++k) {
    float v = ssp * c[k];
    _Float16 hi = (_Float16)v;
    ph.e[k] = hi;
    pl.e[k] = (_Float16)(v - (float)hi);
  }
  size_t row = (size_t)o * KDIM;
  *(uint4*)(wh + row + i * NB) = ph.q;
  *(uint4*)(wl + row + i * NB) = pl.q;
  float sb = scale_base[(size_t)i * OUT_F + o];
  _Float16 hb = (_Float16)sb;
  wh[row + KSPL + i] = hb;
  wl[row + KSPL + i] = (_Float16)(sb - (float)hb);
}

// ---------------------------------------------------------------------------
// Kernel 2: fused basis + GEMM.  y[4096x1024] = A[4096x9216] * W^T
// 128x128 block tile, 256 threads = 8 waves (4 along M x 2 along N),
// each wave owns 2x4 16x16 f32 accumulators. Split-fp16 3-WMMA per tile.
// Software pipeline: A staged in regs one step ahead; B tile moved by
// async global->LDS (double-buffered) when available.
// ---------------------------------------------------------------------------
__global__ __launch_bounds__(256) void kan_gemm(
    const float* __restrict__ x, const float* __restrict__ grid,
    const _Float16* __restrict__ wh, const _Float16* __restrict__ wl,
    float* __restrict__ y) {
#if HAVE_ASYNC_LDS
  __shared__ char lds[6 * LDS_TILE];            // Ah, Al, B[2 bufs][hi|lo]
  char* ldsAh = lds;
  char* ldsAl = lds + LDS_TILE;
  char* ldsB  = lds + 2 * LDS_TILE;             // buf*(2*LDS_TILE) + plane*LDS_TILE
#else
  __shared__ char lds[4 * LDS_TILE];            // Ah, Al, Bh, Bl
  char* ldsAh = lds;
  char* ldsAl = lds + LDS_TILE;
  char* ldsBh = lds + 2 * LDS_TILE;
  char* ldsBl = lds + 3 * LDS_TILE;
#endif

  const int tid  = threadIdx.x;
  const int wave = tid >> 5;
  const int lane = tid & 31;
  const int wm   = wave >> 1;        // 0..3  (M sub-block)
  const int wn   = wave & 1;         // 0..1  (N sub-block)
  const int m16  = lane & 15;
  const int lh   = lane >> 4;        // lane half, selects K chunks per ISA layout

  const int row0 = blockIdx.x * TILE_M;
  const int col0 = blockIdx.y * TILE_N;

  // cooperative tile-load mapping: 2 threads per 64B row, 2x16B chunks each
  const int ldr = tid >> 1;
  const int ldc = (tid & 1) * 2;

  v8f acc[2][4];
  v8f zero = {};
#pragma unroll
  for (int t = 0; t < 2; ++t)
#pragma unroll
    for (int u = 0; u < 4; ++u) acc[t][u] = zero;

  // --- A tile staged in registers one k-step ahead -------------------------
  struct AReg { uint4 h[2], l[2]; int off[2]; } A;

  auto compute_A = [&](int kk) {
    if (kk < KSPL) {
      // Spline region: features kk/8 .. kk/8+3; closed-form uniform cubic
      // B-spline weights (knot origin/step read from the grid input).
      int i0 = kk >> 3;
#pragma unroll
      for (int rep = 0; rep < 2; ++rep) {
        int p = tid + rep * 256;     // 512 (row, feature) pairs
        int f = p & 3;
        int r = p >> 2;
        int i = i0 + f;
        float xv = x[(size_t)(row0 + r) * IN_F + i];
        const float* g = grid + (size_t)i * GRIDP;
        float e3 = g[3];
        float hstep = g[4] - e3;
        float s = (xv - e3) / hstep;
        float cf = floorf(s);
        cf = fminf(fmaxf(cf, 0.f), 4.f);
        int   c = (int)cf;
        float u = s - cf;
        float u2 = u * u, u3 = u2 * u;
        float om = 1.f - u;
        float w0 = om * om * om * (1.f / 6.f);
        float w1 = (3.f * u3 - 6.f * u2 + 4.f) * (1.f / 6.f);
        float w2 = (-3.f * u3 + 3.f * u2 + 3.f * u + 1.f) * (1.f / 6.f);
        float w3 = u3 * (1.f / 6.f);
        union { _Float16 e[8]; uint4 q; } ph, pl;
#pragma unroll
        for (int k = 0; k < 8; ++k) {
          float v = 0.f;
          v = (c == k    ) ? w0 : v;
          v = (c == k - 1) ? w1 : v;
          v = (c == k - 2) ? w2 : v;
          v = (c == k - 3) ? w3 : v;
          _Float16 hi = (_Float16)v;
          ph.e[k] = hi;
          pl.e[k] = (_Float16)(v - (float)hi);
        }
        A.h[rep] = ph.q;
        A.l[rep] = pl.q;
        A.off[rep] = r * APITCH + f * 16;
      }
    } else {
      // Base region: silu(x) for 32 features.
      int i0 = kk - KSPL;
      int r  = tid >> 1;
      int hf = tid & 1;
      union { _Float16 e[16]; uint4 q[2]; } ph, pl;
#pragma unroll
      for (int j = 0; j < 16; ++j) {
        float xv = x[(size_t)(row0 + r) * IN_F + i0 + hf * 16 + j];
        float sv = xv / (1.f + __expf(-xv));
        _Float16 hi = (_Float16)sv;
        ph.e[j] = hi;
        pl.e[j] = (_Float16)(sv - (float)hi);
      }
      A.h[0] = ph.q[0]; A.h[1] = ph.q[1];
      A.l[0] = pl.q[0]; A.l[1] = pl.q[1];
      A.off[0] = r * APITCH + hf * 32;
      A.off[1] = A.off[0] + 16;
    }
  };

  auto store_A = [&]() {
#pragma unroll
    for (int j = 0; j < 2; ++j) {
      *(uint4*)(ldsAh + A.off[j]) = A.h[j];
      *(uint4*)(ldsAl + A.off[j]) = A.l[j];
    }
  };

#if HAVE_ASYNC_LDS
  auto async_B = [&](int kk, int buf) {
    size_t gofs = (size_t)(col0 + ldr) * KDIM + kk + (size_t)ldc * 8;
    char* dh = ldsB + buf * (2 * LDS_TILE) + ldr * APITCH + ldc * 16;
    char* dl = dh + LDS_TILE;
    async_copy_b128(wh + gofs,     dh);
    async_copy_b128(wh + gofs + 8, dh + 16);
    async_copy_b128(wl + gofs,     dl);
    async_copy_b128(wl + gofs + 8, dl + 16);
  };
#else
  uint4 Bh0, Bh1, Bl0, Bl1;
  auto load_B = [&](int kk) {
    size_t gofs = (size_t)(col0 + ldr) * KDIM + kk;
    const uint4* gh = (const uint4*)(wh + gofs) + ldc;
    const uint4* gl = (const uint4*)(wl + gofs) + ldc;
    Bh0 = gh[0]; Bh1 = gh[1];
    Bl0 = gl[0]; Bl1 = gl[1];
  };
  auto store_B = [&]() {
    char* dh = ldsBh + ldr * APITCH + ldc * 16;
    char* dl = ldsBl + ldr * APITCH + ldc * 16;
    *(uint4*)(dh)      = Bh0; *(uint4*)(dh + 16) = Bh1;
    *(uint4*)(dl)      = Bl0; *(uint4*)(dl + 16) = Bl1;
  };
#endif

  auto mma_step = [&](const char* bH, const char* bL) {
    union U { v16h v; uint4 q[2]; };
    U ah[2], al[2], bh[4], bl[4];
#pragma unroll
    for (int t = 0; t < 2; ++t) {
      // 16-bit A 16x32 layout: lane half lh reads K chunks 8*lh and 16+8*lh
      int base = (wm * 32 + t * 16 + m16) * APITCH + lh * 16;
      ah[t].q[0] = *(const uint4*)(ldsAh + base);
      ah[t].q[1] = *(const uint4*)(ldsAh + base + 32);
      al[t].q[0] = *(const uint4*)(ldsAl + base);
      al[t].q[1] = *(const uint4*)(ldsAl + base + 32);
    }
#pragma unroll
    for (int u = 0; u < 4; ++u) {
      // B 32x16 layout: lane = n + 16*lh, holds K chunks 8*lh and 16+8*lh of col n
      int base = (wn * 64 + u * 16 + m16) * APITCH + lh * 16;
      bh[u].q[0] = *(const uint4*)(bH + base);
      bh[u].q[1] = *(const uint4*)(bH + base + 32);
      bl[u].q[0] = *(const uint4*)(bL + base);
      bl[u].q[1] = *(const uint4*)(bL + base + 32);
    }
#pragma unroll
    for (int t = 0; t < 2; ++t)
#pragma unroll
      for (int u = 0; u < 4; ++u) {
        acc[t][u] = __builtin_amdgcn_wmma_f32_16x16x32_f16(
            false, ah[t].v, false, bh[u].v, (short)0, acc[t][u], false, false);
        acc[t][u] = __builtin_amdgcn_wmma_f32_16x16x32_f16(
            false, al[t].v, false, bh[u].v, (short)0, acc[t][u], false, false);
        acc[t][u] = __builtin_amdgcn_wmma_f32_16x16x32_f16(
            false, ah[t].v, false, bl[u].v, (short)0, acc[t][u], false, false);
      }
  };

#if HAVE_ASYNC_LDS
  compute_A(0);
  async_B(0, 0);
  for (int s = 0; s < NSTEPS; ++s) {
    __syncthreads();                         // prior iteration's LDS reads done
    store_A();
    if (s + 1 < NSTEPS) async_B((s + 1) * KSTEP, (s + 1) & 1);
    if (s + 1 < NSTEPS) wait_async_keep4();  // tile s landed; tile s+1 in flight
    else                wait_async_all();
    __syncthreads();                         // A stores + async B(s) visible
    if (s + 1 < NSTEPS) compute_A((s + 1) * KSTEP);  // x loads overlap WMMAs
    const char* bb = ldsB + (s & 1) * (2 * LDS_TILE);
    mma_step(bb, bb + LDS_TILE);
  }
#else
  compute_A(0);
  load_B(0);
  for (int s = 0; s < NSTEPS; ++s) {
    __syncthreads();
    store_A();
    store_B();
    __syncthreads();
    if (s + 1 < NSTEPS) {
      compute_A((s + 1) * KSTEP);            // next tile: loads overlap WMMAs
      load_B((s + 1) * KSTEP);
    }
    mma_step(ldsBh, ldsBl);
  }
#endif

  // C/D 16x16 f32 layout: VGPR r, lanes 0-15 -> (M=r, N=lane);
  // lanes 16-31 -> (M=r+8, N=lane-16)
#pragma unroll
  for (int t = 0; t < 2; ++t)
#pragma unroll
    for (int u = 0; u < 4; ++u)
#pragma unroll
      for (int r = 0; r < 8; ++r) {
        int row = row0 + wm * 32 + t * 16 + r + 8 * lh;
        int col = col0 + wn * 64 + u * 16 + m16;
        y[(size_t)row * OUT_F + col] = acc[t][u][r];
      }
}

// ---------------------------------------------------------------------------
// Kernel 3: row LayerNorm.  One 256-thread block per batch row.
// ---------------------------------------------------------------------------
__global__ __launch_bounds__(256) void layernorm_row(
    const float* __restrict__ y, const float* __restrict__ gamma,
    const float* __restrict__ beta, float* __restrict__ out) {
  __shared__ float red0[8];
  __shared__ float red1[8];
  int b = blockIdx.x;
  const float* row = y + (size_t)b * OUT_F;
  float v[4];
  float s = 0.f, s2 = 0.f;
#pragma unroll
  for (int j = 0; j < 4; ++j) {
    v[j] = row[threadIdx.x + j * 256];
    s += v[j];
    s2 += v[j] * v[j];
  }
#pragma unroll
  for (int off = 16; off > 0; off >>= 1) {
    s  += __shfl_down(s, off, 32);
    s2 += __shfl_down(s2, off, 32);
  }
  int wv = threadIdx.x >> 5, lane = threadIdx.x & 31;
  if (lane == 0) { red0[wv] = s; red1[wv] = s2; }
  __syncthreads();
  float ts = 0.f, ts2 = 0.f;
#pragma unroll
  for (int w = 0; w < 8; ++w) { ts += red0[w]; ts2 += red1[w]; }
  float mu  = ts * (1.f / OUT_F);
  float var = ts2 * (1.f / OUT_F) - mu * mu;
  float inv = rsqrtf(var + LN_EPS);
#pragma unroll
  for (int j = 0; j < 4; ++j) {
    int o = threadIdx.x + j * 256;
    out[(size_t)b * OUT_F + o] = (v[j] - mu) * inv * gamma[o] + beta[o];
  }
}

// ---------------------------------------------------------------------------
extern "C" void kernel_launch(void* const* d_in, const int* in_sizes, int n_in,
                              void* d_out, int out_size, void* d_ws, size_t ws_size,
                              hipStream_t stream) {
  (void)in_sizes; (void)n_in; (void)out_size; (void)ws_size;
  const float* x          = (const float*)d_in[0];
  const float* coef       = (const float*)d_in[1];
  const float* scale_base = (const float*)d_in[2];
  const float* scale_sp   = (const float*)d_in[3];
  const float* gamma      = (const float*)d_in[4];
  const float* beta       = (const float*)d_in[5];
  const float* grid       = (const float*)d_in[6];

  char* ws = (char*)d_ws;
  _Float16* wh = (_Float16*)(ws + WH_OFF);
  _Float16* wl = (_Float16*)(ws + WL_OFF);
  float*    y  = (float*)(ws + Y_OFF);
  float*    out = (float*)d_out;

  pack_w<<<(IN_F * OUT_F + 255) / 256, 256, 0, stream>>>(coef, scale_base, scale_sp, wh, wl);
  dim3 g(BATCH / TILE_M, OUT_F / TILE_N);
  kan_gemm<<<g, 256, 0, stream>>>(x, grid, wh, wl, y);
  layernorm_row<<<BATCH, 256, 0, stream>>>(y, gamma, beta, out);
}
